// PowerNetLayer_14912126452490
// MI455X (gfx1250) — compile-verified, matrix operand order checked by reading, not run on hardware
//
#include <hip/hip_runtime.h>
#include <hip/hip_bf16.h>

// ---------------------------------------------------------------------------
// PowerNet GNN layer for MI455X (gfx1250, wave32, WMMA).
// All GEMMs run on v_wmma_f32_16x16x32_bf16 (fp32 accumulate).
// Block tile: 64 rows x 256 cols; 16 waves; each wave owns a 16-col slab and
// 4 row-tiles so every weight (B) fragment feeds 4 WMMAs.
// ---------------------------------------------------------------------------

typedef __attribute__((ext_vector_type(16))) __bf16 bf16x16;
typedef __attribute__((ext_vector_type(8)))  float  f32x8;

static __device__ __forceinline__ __bf16 to_bf16(float f) { return (__bf16)f; }

struct Seg {
    const float* ptr;   // row-major source, row stride = stride floats
    const int*   idx;   // gather index (nullptr = identity)
    int          stride;
    int          coff;  // column offset inside source row
};

struct MlpArgs {
    int   n;            // number of rows
    Seg   seg[4];       // FIN/256 segments used
    const __bf16* w1t;  // [256][FIN]  = W1^T, bf16
    const float*  b1;   // [256]
    const __bf16* w2t;  // [256][256]  = W2^T, bf16
    const float*  b2;   // [256]
    float*        out;       // [n][256] (nullptr = no dense store)
    const float*  res;       // residual, identity-indexed [n][256] (nullptr = none)
    float*        scat;      // scatter-add target (nullptr = none)
    const int*    scat_idx;  // scatter row index (nullptr = identity)
    int           scat_stride;
    int           scat_coff;
};

#define ROWS 64   // rows per block  (= 16 * RT)
#define RT   4    // 16-row tiles per wave

// Fused gather -> (X @ W1 + b1).relu -> (@ W2 + b2).relu -> {store+residual, scatter-add}
template <int FIN>
__global__ __launch_bounds__(512) void mlp2_kernel(MlpArgs A)
{
    constexpr int NSEG = FIN / 256;
    __shared__ __align__(32) __bf16 sx[ROWS * FIN];   // bf16 staged input tile
    __shared__ __align__(32) __bf16 sh[ROWS * 256];   // bf16 hidden tile

    const int row0 = blockIdx.x * ROWS;
    const int tid  = threadIdx.x;

    // ---- stage 0: gather + concat + fp32->bf16 into LDS (coalesced 256-wide) ----
#pragma unroll
    for (int s = 0; s < NSEG; ++s) {
        const float* p  = A.seg[s].ptr;
        const int*   ix = A.seg[s].idx;
        const int    st = A.seg[s].stride;
        const int    co = A.seg[s].coff;
        for (int e = tid; e < ROWS * 256; e += 512) {
            const int r = e >> 8, c = e & 255;
            const int g = row0 + r;
            float v = 0.0f;
            if (g < A.n) {
                const int sr = ix ? ix[g] : g;
                v = p[(size_t)sr * st + co + c];
            }
            sx[r * FIN + s * 256 + c] = to_bf16(v);
        }
    }
    __syncthreads();

    const int lane = tid & 31;
    const int wv   = tid >> 5;        // 0..15
    const int cb   = wv << 4;         // column base of this wave's slab
    const int half = lane >> 4;       // 0: K/N 0..15 half, 1: 16..31 half
    const int ln   = lane & 15;

    f32x8 acc[RT];

    // ---- layer 1: H = relu(X @ W1 + b1), K = FIN ----
    {
        const float b = A.b1[cb + ln];
#pragma unroll
        for (int t = 0; t < RT; ++t)
#pragma unroll
            for (int i = 0; i < 8; ++i) acc[t][i] = b;
    }
    {
        const __bf16* brow = A.w1t + (size_t)(cb + ln) * FIN + half * 16;
        const __bf16* abase = sx + (size_t)ln * FIN + half * 16;
#pragma unroll 4
        for (int kb = 0; kb < FIN; kb += 32) {
            bf16x16 bfr = *(const bf16x16*)(brow + kb);
#pragma unroll
            for (int t = 0; t < RT; ++t) {
                bf16x16 af = *(const bf16x16*)(abase + (size_t)(16 * t) * FIN + kb);
                acc[t] = __builtin_amdgcn_wmma_f32_16x16x32_bf16(
                    false, af, false, bfr, (short)0, acc[t], false, false);
            }
        }
    }

    // relu + bf16 convert into LDS hidden tile
#pragma unroll
    for (int t = 0; t < RT; ++t)
#pragma unroll
        for (int i = 0; i < 8; ++i) {
            const float v = acc[t][i] > 0.0f ? acc[t][i] : 0.0f;
            sh[(16 * t + i + half * 8) * 256 + cb + ln] = to_bf16(v);
        }
    __syncthreads();

    // ---- layer 2: Y = relu(H @ W2 + b2), K = 256 ----
    {
        const float b = A.b2[cb + ln];
#pragma unroll
        for (int t = 0; t < RT; ++t)
#pragma unroll
            for (int i = 0; i < 8; ++i) acc[t][i] = b;
    }
    {
        const __bf16* brow = A.w2t + (size_t)(cb + ln) * 256 + half * 16;
        const __bf16* abase = sh + (size_t)ln * 256 + half * 16;
#pragma unroll 4
        for (int kb = 0; kb < 256; kb += 32) {
            bf16x16 bfr = *(const bf16x16*)(brow + kb);
#pragma unroll
            for (int t = 0; t < RT; ++t) {
                bf16x16 af = *(const bf16x16*)(abase + (size_t)(16 * t) * 256 + kb);
                acc[t] = __builtin_amdgcn_wmma_f32_16x16x32_bf16(
                    false, af, false, bfr, (short)0, acc[t], false, false);
            }
        }
    }

    // ---- epilogue: relu, residual store, fused segment-sum scatter ----
    const int col = cb + ln;
#pragma unroll
    for (int t = 0; t < RT; ++t) {
#pragma unroll
        for (int i = 0; i < 8; ++i) {
            const int m = 16 * t + i + half * 8;
            const int g = row0 + m;
            if (g >= A.n) continue;
            const float v = acc[t][i] > 0.0f ? acc[t][i] : 0.0f;
            if (A.out) {
                float o = v;
                if (A.res) o += A.res[(size_t)g * 256 + col];
                A.out[(size_t)g * 256 + col] = o;
            }
            if (A.scat) {
                const int tr = A.scat_idx ? A.scat_idx[g] : g;
                __hip_atomic_fetch_add(
                    A.scat + (size_t)tr * A.scat_stride + A.scat_coff + col,
                    v, __ATOMIC_RELAXED, __HIP_MEMORY_SCOPE_AGENT);
            }
        }
    }
}

// src: K x N row-major fp32  ->  dst: N x K row-major bf16 (transposed + converted)
__global__ void transpose_cvt_kernel(const float* __restrict__ src,
                                     __bf16* __restrict__ dst, int K, int N)
{
    const long i = (long)blockIdx.x * 256 + threadIdx.x;
    if (i < (long)K * N) {
        const int n = (int)(i / K);
        const int k = (int)(i % K);
        dst[i] = (__bf16)src[(long)k * N + n];
    }
}

// ---------------------------------------------------------------------------

extern "C" void kernel_launch(void* const* d_in, const int* in_sizes, int n_in,
                              void* d_out, int out_size, void* d_ws, size_t ws_size,
                              hipStream_t stream)
{
    (void)n_in; (void)out_size; (void)ws_size;
    constexpr int H = 256;

    // ---- inputs (setup_inputs dict order; params nested dict in insertion order) ----
    const float* bus         = (const float*)d_in[0];
    const float* shunt       = (const float*)d_in[1];
    const float* gen         = (const float*)d_in[2];
    const float* load        = (const float*)d_in[3];
    const float* branch_attr = (const float*)d_in[4];
    const int NB = in_sizes[0] / H;
    const int NS = in_sizes[1] / H;
    const int NG = in_sizes[2] / H;
    const int NL = in_sizes[3] / H;
    const int NE = in_sizes[4] / H;
    const int*   shunt_to_bus = (const int*)d_in[33];
    const int*   gen_to_bus   = (const int*)d_in[34];
    const int*   load_to_bus  = (const int*)d_in[35];
    const int*   branch_src   = (const int*)d_in[36];        // branch_index[0]
    const int*   branch_dst   = ((const int*)d_in[36]) + NE; // branch_index[1]

    // subnet order: shunt, gen, load, bus, branch, bus_and_branch, bus_and_neighbours
    const int fins[7] = {2 * H, 2 * H, 2 * H, 4 * H, 3 * H, 2 * H, 2 * H};
    const float* W1[7]; const float* B1[7]; const float* W2[7]; const float* B2[7];
    for (int i = 0; i < 7; ++i) {
        W1[i] = (const float*)d_in[5 + 4 * i + 0];
        B1[i] = (const float*)d_in[5 + 4 * i + 1];
        W2[i] = (const float*)d_in[5 + 4 * i + 2];
        B2[i] = (const float*)d_in[5 + 4 * i + 3];
    }

    // ---- workspace layout ----
    char* ws = (char*)d_ws;
    size_t off = 0;
    __bf16* w1t[7]; __bf16* w2t[7];
    for (int i = 0; i < 7; ++i) { w1t[i] = (__bf16*)(ws + off); off += (size_t)H * fins[i] * 2; }
    for (int i = 0; i < 7; ++i) { w2t[i] = (__bf16*)(ws + off); off += (size_t)H * H * 2; }
    float* segacc = (float*)(ws + off); off += (size_t)NB * (3 * H) * 4; // [seg_shunt|seg_gen|seg_load]
    float* bus1   = (float*)(ws + off); off += (size_t)NB * H * 4;
    float* bneigh = (float*)(ws + off); off += (size_t)NB * H * 4;

    hipMemsetAsync(segacc, 0, (size_t)NB * (3 * H) * 4, stream);
    hipMemsetAsync(bneigh, 0, (size_t)NB * H * 4, stream);

    // ---- weight prep: transpose + convert to bf16 (tiny, every call) ----
    for (int i = 0; i < 7; ++i) {
        long n1 = (long)H * fins[i];
        transpose_cvt_kernel<<<(int)((n1 + 255) / 256), 256, 0, stream>>>(W1[i], w1t[i], fins[i], H);
        long n2 = (long)H * H;
        transpose_cvt_kernel<<<(int)((n2 + 255) / 256), 256, 0, stream>>>(W2[i], w2t[i], H, H);
    }

    // ---- output sections ----
    float* out = (float*)d_out;
    float* o_bus    = out;
    float* o_shunt  = o_bus + (size_t)NB * H;
    float* o_gen    = o_shunt + (size_t)NS * H;
    float* o_load   = o_gen + (size_t)NG * H;
    float* o_branch = o_load + (size_t)NL * H;

    auto grid = [](int n) { return dim3((unsigned)((n + ROWS - 1) / ROWS)); };

    // 1) shunt MLP: [shunt, bus[s2b]] -> out + residual(shunt); scatter into segacc[:,0:256]
    {
        MlpArgs a{}; a.n = NS;
        a.seg[0] = {shunt, nullptr, H, 0};
        a.seg[1] = {bus, shunt_to_bus, H, 0};
        a.w1t = w1t[0]; a.b1 = B1[0]; a.w2t = w2t[0]; a.b2 = B2[0];
        a.out = o_shunt; a.res = shunt;
        a.scat = segacc; a.scat_idx = shunt_to_bus; a.scat_stride = 3 * H; a.scat_coff = 0;
        mlp2_kernel<512><<<grid(NS), 512, 0, stream>>>(a);
    }
    // 2) gen MLP
    {
        MlpArgs a{}; a.n = NG;
        a.seg[0] = {gen, nullptr, H, 0};
        a.seg[1] = {bus, gen_to_bus, H, 0};
        a.w1t = w1t[1]; a.b1 = B1[1]; a.w2t = w2t[1]; a.b2 = B2[1];
        a.out = o_gen; a.res = gen;
        a.scat = segacc; a.scat_idx = gen_to_bus; a.scat_stride = 3 * H; a.scat_coff = H;
        mlp2_kernel<512><<<grid(NG), 512, 0, stream>>>(a);
    }
    // 3) load MLP
    {
        MlpArgs a{}; a.n = NL;
        a.seg[0] = {load, nullptr, H, 0};
        a.seg[1] = {bus, load_to_bus, H, 0};
        a.w1t = w1t[2]; a.b1 = B1[2]; a.w2t = w2t[2]; a.b2 = B2[2];
        a.out = o_load; a.res = load;
        a.scat = segacc; a.scat_idx = load_to_bus; a.scat_stride = 3 * H; a.scat_coff = 2 * H;
        mlp2_kernel<512><<<grid(NL), 512, 0, stream>>>(a);
    }
    // 4) bus MLP: [bus, seg_shunt, seg_gen, seg_load] -> bus1 (ws)
    {
        MlpArgs a{}; a.n = NB;
        a.seg[0] = {bus, nullptr, H, 0};
        a.seg[1] = {segacc, nullptr, 3 * H, 0};
        a.seg[2] = {segacc, nullptr, 3 * H, H};
        a.seg[3] = {segacc, nullptr, 3 * H, 2 * H};
        a.w1t = w1t[3]; a.b1 = B1[3]; a.w2t = w2t[3]; a.b2 = B2[3];
        a.out = bus1;
        mlp2_kernel<1024><<<grid(NB), 512, 0, stream>>>(a);
    }
    // 5) branch MLP: [bus1[src], branch_attr, bus1[dst]] -> out + residual(branch_attr)
    {
        MlpArgs a{}; a.n = NE;
        a.seg[0] = {bus1, branch_src, H, 0};
        a.seg[1] = {branch_attr, nullptr, H, 0};
        a.seg[2] = {bus1, branch_dst, H, 0};
        a.w1t = w1t[4]; a.b1 = B1[4]; a.w2t = w2t[4]; a.b2 = B2[4];
        a.out = o_branch; a.res = branch_attr;
        mlp2_kernel<768><<<grid(NE), 512, 0, stream>>>(a);
    }
    // 6) bus_and_branch MLP: [bus1[dst], branch_attr] -> scatter-add into bneigh via src
    {
        MlpArgs a{}; a.n = NE;
        a.seg[0] = {bus1, branch_dst, H, 0};
        a.seg[1] = {branch_attr, nullptr, H, 0};
        a.w1t = w1t[5]; a.b1 = B1[5]; a.w2t = w2t[5]; a.b2 = B2[5];
        a.scat = bneigh; a.scat_idx = branch_src; a.scat_stride = H; a.scat_coff = 0;
        mlp2_kernel<512><<<grid(NE), 512, 0, stream>>>(a);
    }
    // 7) bus_and_neighbours MLP: [bneigh, bus1] -> out + residual(original bus)
    {
        MlpArgs a{}; a.n = NB;
        a.seg[0] = {bneigh, nullptr, H, 0};
        a.seg[1] = {bus1, nullptr, H, 0};
        a.w1t = w1t[6]; a.b1 = B1[6]; a.w2t = w2t[6]; a.b2 = B2[6];
        a.out = o_bus; a.res = bus;
        mlp2_kernel<512><<<grid(NB), 512, 0, stream>>>(a);
    }
}
